// MultiHeadAttention_36687610643098
// MI455X (gfx1250) — matrix operand order repaired and checked by reading, not compiled, not run on hardware
//
#include <hip/hip_runtime.h>
#include <stdint.h>

// ---------------------------------------------------------------------------
// MHA forward for B=4, T=2048, D=1024, H=16, HD=64 on gfx1250 (wave32, WMMA).
// d_out = out (B*T*D fp32) ++ weights (B*H*T,T fp32).
// Key layout rule: WMMA B-operands want (N-major, K-contiguous) storage ->
// weights packed column-major bf16, V stored column-major (hd, t) bf16.
// Fragments then load as aligned b128s (global and LDS) with no shuffles.
// ---------------------------------------------------------------------------

#define BB 4
#define TT 2048
#define DD 1024
#define HH 16
#define HDD 64

typedef __attribute__((ext_vector_type(16))) __bf16 v16bf;
typedef __attribute__((ext_vector_type(8)))  float  v8f;

union ABFrag {
  v16bf v;
  unsigned short s[16];
  unsigned int   u[8];
  uint4          q[2];
};

__device__ __forceinline__ unsigned short f32_to_bf16(float f) {
  union { float f; unsigned int u; } x;
  x.f = f;
  unsigned int r = x.u + 0x7FFFu + ((x.u >> 16) & 1u);  // RNE
  return (unsigned short)(r >> 16);
}

__device__ __forceinline__ v8f wmma_bf16(const ABFrag& a, const ABFrag& b, v8f c) {
  return __builtin_amdgcn_wmma_f32_16x16x32_bf16(false, a.v, false, b.v, (short)0, c, false, false);
}

// 16x32 bf16 A-fragment from a row-major bf16 row at col kk (two b128 loads).
__device__ __forceinline__ void load_a_frag(ABFrag& a, const unsigned short* row,
                                            int kk, int half) {
  a.q[0] = *(const uint4*)(row + kk + half * 8);
  a.q[1] = *(const uint4*)(row + kk + 16 + half * 8);
}

// 32x16 bf16 B-fragment from (N-major, K-contiguous) storage: col base `colp`
// (bf16*, already offset to this column's K run at kk) -> two b128 loads.
__device__ __forceinline__ void load_b_frag(ABFrag& b, const unsigned short* colp,
                                            int half) {
  b.q[0] = *(const uint4*)(colp + half * 8);
  b.q[1] = *(const uint4*)(colp + 16 + half * 8);
}

// ---------------------------------------------------------------------------
__global__ __launch_bounds__(256)
void cvt_bf16_kernel(const float* __restrict__ src, unsigned short* __restrict__ dst,
                     long total) {
  for (long i = (long)blockIdx.x * blockDim.x + threadIdx.x; i < total;
       i += (long)gridDim.x * blockDim.x)
    dst[i] = f32_to_bf16(src[i]);
}

// ---------------------------------------------------------------------------
// Transpose-pack: nmat fp32 (K,N) row-major mats -> bf16 (N,K) column-major.
// Output-coalesced iteration.
// ---------------------------------------------------------------------------
__global__ __launch_bounds__(256)
void pack_cm_kernel(const float* __restrict__ src, unsigned short* __restrict__ dst,
                    int K, int N, int nmat) {
  long total = (long)nmat * K * N;
  for (long i = (long)blockIdx.x * blockDim.x + threadIdx.x; i < total;
       i += (long)gridDim.x * blockDim.x) {
    long m = i / ((long)K * N);
    long r = i - m * (long)K * N;
    int  n = (int)(r / K);
    int  k = (int)(r - (long)n * K);
    dst[i] = f32_to_bf16(src[m * (long)K * N + (long)k * N + n]);
  }
}

// ---------------------------------------------------------------------------
// QKV projection. Block = 128 threads (4 waves), each wave a 16x64 tile.
// W K-step tile (64 cols x 32 K bf16 = 4 KB) async-staged to LDS, double
// buffered; LDS layout = 64 x 64B rows so fragments are 2x ds_load_b128.
// ---------------------------------------------------------------------------
__global__ __launch_bounds__(128)
void qkv_gemm_kernel(const unsigned short* __restrict__ Xbf,  // (B*T, D) bf16
                     const unsigned short* __restrict__ Wcm,  // 48 mats (HD, D) cm bf16
                     const float* __restrict__ bq,
                     const float* __restrict__ bk,
                     const float* __restrict__ bv,
                     unsigned short* __restrict__ Qbf,   // (B,H,T,HD)
                     unsigned short* __restrict__ Kbf,   // (B,H,T,HD)
                     unsigned short* __restrict__ Vcm) { // (B,H,HD,T)
  __shared__ __align__(16) unsigned int ldsW[2][1024];  // 2 x 4 KB

  const int mtile = blockIdx.x;
  const int ymat  = blockIdx.y;        // 0..47: mat*16 + head
  const int mat   = ymat / HH;
  const int h     = ymat % HH;
  const int wave  = threadIdx.x >> 5;
  const int lane  = threadIdx.x & 31;
  const int half  = lane >> 4;
  const int l16   = lane & 15;
  const int mbase = mtile * 64 + wave * 16;

  const unsigned short* W = Wcm + (long)ymat * HDD * DD;  // 64 rows x 1024 bf16
  const unsigned long long wbase = (unsigned long long)(uintptr_t)W;
  const unsigned short* Arow = Xbf + (long)(mbase + l16) * DD;

  // Copy tile for K-step kk: for each col n<64, 64B run W[n*D + kk ..].
  // 256 b128 chunks: n = c>>2, part = c&3. Thread issues c=tid, c=tid+128.
  auto issue_tile = [&](int kk, int buf) {
#pragma unroll
    for (int it = 0; it < 2; ++it) {
      const int c    = threadIdx.x + it * 128;
      const int n    = c >> 2;
      const int part = c & 3;
      const unsigned int ldsoff =
          (unsigned int)(uintptr_t)(&ldsW[buf][0]) + (unsigned int)(n * 64 + part * 16);
      const unsigned int goff =
          (unsigned int)(((unsigned)n * DD + (unsigned)kk) * 2u + (unsigned)part * 16u);
      asm volatile("global_load_async_to_lds_b128 %0, %1, %2"
                   :: "v"(ldsoff), "v"(goff), "s"(wbase) : "memory");
    }
  };

  const v8f vz = {0.f, 0.f, 0.f, 0.f, 0.f, 0.f, 0.f, 0.f};
  v8f acc[4] = {vz, vz, vz, vz};

  issue_tile(0, 0);
  for (int kk = 0; kk < DD; kk += 32) {
    const int buf = (kk >> 5) & 1;
    if (kk + 32 < DD) {
      issue_tile(kk + 32, buf ^ 1);
      asm volatile("s_wait_asynccnt 0x2" ::: "memory");  // current tile landed
    } else {
      asm volatile("s_wait_asynccnt 0x0" ::: "memory");
    }
    __syncthreads();

    __builtin_prefetch(Arow + kk + 64, 0, 1);

    ABFrag a;
    load_a_frag(a, Arow, kk, half);
#pragma unroll
    for (int nt = 0; nt < 4; ++nt) {
      const int n = nt * 16 + l16;
      ABFrag bf;
      load_b_frag(bf, (const unsigned short*)&ldsW[buf][n * 16], half);
      acc[nt] = wmma_bf16(a, bf, acc[nt]);
    }
    __syncthreads();  // all waves done reading buf before overwrite
  }

  const float* bias = ((mat == 0) ? bq : (mat == 1) ? bk : bv) + h * HDD;
#pragma unroll
  for (int nt = 0; nt < 4; ++nt) {
    const int n  = nt * 16 + l16;
    const float bb = bias[n];
#pragma unroll
    for (int r = 0; r < 8; ++r) {
      const int   m   = mbase + half * 8 + r;
      const unsigned short bfv = f32_to_bf16(acc[nt][r] + bb);
      const int  b    = m / TT;
      const int  t    = m - b * TT;
      const long base = ((long)(b * HH + h)) * TT * HDD;
      if (mat == 0)      Qbf[base + (long)t * HDD + n] = bfv;
      else if (mat == 1) Kbf[base + (long)t * HDD + n] = bfv;
      else               Vcm[base + (long)n * TT + t] = bfv;  // column-major V
    }
  }
}

// ---------------------------------------------------------------------------
// Attention: one block per (16-query tile, head, batch). 8 waves.
//  P1: wave w computes score strip S[0:16][w*256..+256) via WMMA.
//  P2: row softmax in LDS; normalized P streamed to weights with NT stores.
//  P3: out_h = P @ V (V column-major -> b128 B-fragments); LDS reduction.
// Dynamic LDS: S(16*2048 f32) + part(8*16*64 f32) + red(256) + rsum(16).
// ---------------------------------------------------------------------------
__global__ __launch_bounds__(256)
void attn_kernel(const unsigned short* __restrict__ Qbf,
                 const unsigned short* __restrict__ Kbf,
                 const unsigned short* __restrict__ Vcm,
                 float* __restrict__ weights_out,        // (B,H,T,T)
                 unsigned short* __restrict__ concat) {  // (B,T,D) bf16
  extern __shared__ float smem[];
  float* S    = smem;                    // 16*2048
  float* part = S + 16 * TT;             // 8*16*64
  float* red  = part + 8 * 16 * HDD;     // 256
  float* rsum = red + 256;               // 16

  const int qtile = blockIdx.x;
  const int h     = blockIdx.y;
  const int b     = blockIdx.z;
  const int wave  = threadIdx.x >> 5;
  const int lane  = threadIdx.x & 31;
  const int half  = lane >> 4;
  const int l16   = lane & 15;
  const long bh   = (long)(b * HH + h);
  const int qbase = qtile * 16;

  const unsigned short* Q = Qbf + bh * TT * HDD;
  const unsigned short* K = Kbf + bh * TT * HDD;
  const unsigned short* V = Vcm + bh * TT * HDD;

  // ---- Phase 1: scores -------------------------------------------------
  ABFrag aq[2];
  {
    const unsigned short* Qrow = Q + (long)(qbase + l16) * HDD;
    load_a_frag(aq[0], Qrow, 0, half);
    load_a_frag(aq[1], Qrow, 32, half);
  }
  const v8f vz = {0.f, 0.f, 0.f, 0.f, 0.f, 0.f, 0.f, 0.f};
  const float inv_scale = 0.125f;  // 1/sqrt(64)

  for (int nt = 0; nt < 16; ++nt) {
    const int keyb = wave * 256 + nt * 16;
    const int key  = keyb + l16;
    const unsigned short* Krow = K + (long)key * HDD;  // K^T col = key row
    v8f acc = vz;
#pragma unroll
    for (int ks = 0; ks < 2; ++ks) {
      ABFrag bk;
      load_b_frag(bk, Krow + ks * 32, half);
      acc = wmma_bf16(aq[ks], bk, acc);
    }
#pragma unroll
    for (int r = 0; r < 8; ++r)
      S[(half * 8 + r) * TT + key] = acc[r] * inv_scale;
  }
  __syncthreads();

  // ---- Phase 2: softmax + NT weight store ------------------------------
  const int row = threadIdx.x >> 4;
  const int sub = threadIdx.x & 15;
  float* Srow = S + row * TT;

  float mx = -3.4e38f;
  for (int i = sub; i < TT; i += 16) mx = fmaxf(mx, Srow[i]);
  red[row * 16 + sub] = mx;
  __syncthreads();
  mx = -3.4e38f;
#pragma unroll
  for (int i = 0; i < 16; ++i) mx = fmaxf(mx, red[row * 16 + i]);

  float sum = 0.f;
  for (int i = sub; i < TT; i += 16) {
    const float e = __expf(Srow[i] - mx);
    Srow[i] = e;
    sum += e;
  }
  __syncthreads();
  red[row * 16 + sub] = sum;
  __syncthreads();
  float tsum = 0.f;
#pragma unroll
  for (int i = 0; i < 16; ++i) tsum += red[row * 16 + i];
  if (sub == 0) rsum[row] = 1.0f / tsum;
  __syncthreads();

  float* wout = weights_out + (bh * TT + qbase) * (long)TT;
  for (int i = threadIdx.x; i < 16 * TT; i += 256) {
    const int r = i >> 11;           // T == 2048
    const int c = i & (TT - 1);
    const float p = S[r * TT + c] * rsum[r];
    S[r * TT + c] = p;
    __builtin_nontemporal_store(p, wout + (long)r * TT + c);  // 1 GB stream
  }
  __syncthreads();

  // ---- Phase 3: out_h = P @ V ------------------------------------------
  v8f acc[4] = {vz, vz, vz, vz};
  for (int ks = 0; ks < 8; ++ks) {
    const int kb = wave * 256 + ks * 32;
    ABFrag ap;
    const float* Sr = S + l16 * TT + kb;
#pragma unroll
    for (int j = 0; j < 8; ++j) {
      ap.s[j]     = f32_to_bf16(Sr[half * 8 + j]);
      ap.s[8 + j] = f32_to_bf16(Sr[16 + half * 8 + j]);
    }
#pragma unroll
    for (int nt = 0; nt < 4; ++nt) {
      const int n = nt * 16 + l16;
      ABFrag bv;
      load_b_frag(bv, V + (long)n * TT + kb, half);  // column-major V
      acc[nt] = wmma_bf16(ap, bv, acc[nt]);
    }
  }
#pragma unroll
  for (int nt = 0; nt < 4; ++nt) {
    const int n = nt * 16 + l16;
#pragma unroll
    for (int r = 0; r < 8; ++r)
      part[(wave * 16 + half * 8 + r) * HDD + n] = acc[nt][r];
  }
  __syncthreads();

  for (int i = threadIdx.x; i < 16 * HDD; i += 256) {
    const int q = i >> 6;
    const int n = i & 63;
    float s = 0.f;
#pragma unroll
    for (int w = 0; w < 8; ++w) s += part[(w * 16 + q) * HDD + n];
    concat[((long)b * TT + qbase + q) * DD + h * HDD + n] = f32_to_bf16(s);
  }
}

// ---------------------------------------------------------------------------
// Output projection: out = concat (8192x1024 bf16) @ Wo + bo.
// Wo column-major; 64-col x 32-K tile (4 KB) async-staged per K-step.
// ---------------------------------------------------------------------------
__global__ __launch_bounds__(128)
void oproj_kernel(const unsigned short* __restrict__ A,    // (B*T, D) bf16
                  const unsigned short* __restrict__ WoCm, // (D, D) column-major
                  const float* __restrict__ bo,
                  float* __restrict__ out) {               // (B*T, D) fp32
  __shared__ __align__(16) unsigned int ldsW[2][1024];

  const int mtile = blockIdx.x;
  const int ntile = blockIdx.y;     // 16 slabs of 64 cols
  const int wave  = threadIdx.x >> 5;
  const int lane  = threadIdx.x & 31;
  const int half  = lane >> 4;
  const int l16   = lane & 15;
  const int mbase = mtile * 64 + wave * 16;
  const int nbase = ntile * 64;

  const unsigned long long wbase = (unsigned long long)(uintptr_t)WoCm;
  const unsigned short* Arow = A + (long)(mbase + l16) * DD;

  auto issue_tile = [&](int kk, int buf) {
#pragma unroll
    for (int it = 0; it < 2; ++it) {
      const int c    = threadIdx.x + it * 128;
      const int n    = c >> 2;          // local col 0..63
      const int part = c & 3;
      const unsigned int ldsoff =
          (unsigned int)(uintptr_t)(&ldsW[buf][0]) + (unsigned int)(n * 64 + part * 16);
      const unsigned int goff =
          (unsigned int)(((unsigned)(nbase + n) * DD + (unsigned)kk) * 2u +
                         (unsigned)part * 16u);
      asm volatile("global_load_async_to_lds_b128 %0, %1, %2"
                   :: "v"(ldsoff), "v"(goff), "s"(wbase) : "memory");
    }
  };

  const v8f vz = {0.f, 0.f, 0.f, 0.f, 0.f, 0.f, 0.f, 0.f};
  v8f acc[4] = {vz, vz, vz, vz};

  issue_tile(0, 0);
  for (int kk = 0; kk < DD; kk += 32) {
    const int buf = (kk >> 5) & 1;
    if (kk + 32 < DD) {
      issue_tile(kk + 32, buf ^ 1);
      asm volatile("s_wait_asynccnt 0x2" ::: "memory");
    } else {
      asm volatile("s_wait_asynccnt 0x0" ::: "memory");
    }
    __syncthreads();

    __builtin_prefetch(Arow + kk + 64, 0, 1);

    ABFrag a;
    load_a_frag(a, Arow, kk, half);
#pragma unroll
    for (int nt = 0; nt < 4; ++nt) {
      ABFrag bf;
      load_b_frag(bf, (const unsigned short*)&ldsW[buf][(nt * 16 + l16) * 16], half);
      acc[nt] = wmma_bf16(a, bf, acc[nt]);
    }
    __syncthreads();
  }

#pragma unroll
  for (int nt = 0; nt < 4; ++nt) {
    const int n  = nbase + nt * 16 + l16;
    const float bb = bo[n];
#pragma unroll
    for (int r = 0; r < 8; ++r)
      out[(long)(mbase + half * 8 + r) * DD + n] = acc[nt][r] + bb;
  }
}

// ---------------------------------------------------------------------------
extern "C" void kernel_launch(void* const* d_in, const int* in_sizes, int n_in,
                              void* d_out, int out_size, void* d_ws, size_t ws_size,
                              hipStream_t stream) {
  (void)in_sizes; (void)n_in; (void)out_size; (void)ws_size;

  const float* X  = (const float*)d_in[0];
  const float* Wq = (const float*)d_in[1];
  const float* bq = (const float*)d_in[2];
  const float* Wk = (const float*)d_in[3];
  const float* bk = (const float*)d_in[4];
  const float* Wv = (const float*)d_in[5];
  const float* bv = (const float*)d_in[6];
  const float* Wo = (const float*)d_in[7];
  const float* bo = (const float*)d_in[8];

  float* out     = (float*)d_out;                       // (B,T,D)
  float* weights = out + (long)BB * TT * DD;            // (B,H,T,T)

  // Workspace carve-up (~88 MB total).
  char* ws = (char*)d_ws;
  size_t off = 0;
  const size_t qkv_elems = (size_t)BB * HH * TT * HDD;  // 8,388,608
  unsigned short* Qbf  = (unsigned short*)(ws + off); off += qkv_elems * 2;
  unsigned short* Kbf  = (unsigned short*)(ws + off); off += qkv_elems * 2;
  unsigned short* Vcm  = (unsigned short*)(ws + off); off += qkv_elems * 2;
  unsigned short* Xbf  = (unsigned short*)(ws + off); off += (size_t)BB * TT * DD * 2;
  unsigned short* Conc = (unsigned short*)(ws + off); off += (size_t)BB * TT * DD * 2;
  unsigned short* Wqkv = (unsigned short*)(ws + off); off += (size_t)3 * HH * DD * HDD * 2;
  unsigned short* WoCm = (unsigned short*)(ws + off); off += (size_t)DD * DD * 2;

  const long per = (long)HH * DD * HDD;

  // Stage 0: weight transpose-pack (column-major bf16) + X bf16 conversion.
  pack_cm_kernel<<<1024, 256, 0, stream>>>(Wq, Wqkv,           DD, HDD, HH);
  pack_cm_kernel<<<1024, 256, 0, stream>>>(Wk, Wqkv + per,     DD, HDD, HH);
  pack_cm_kernel<<<1024, 256, 0, stream>>>(Wv, Wqkv + 2 * per, DD, HDD, HH);
  pack_cm_kernel<<<1024, 256, 0, stream>>>(Wo, WoCm,           DD, DD, 1);
  cvt_bf16_kernel<<<4096, 256, 0, stream>>>(X, Xbf, (long)BB * TT * DD);

  // Stage 1: QKV projections.
  dim3 g1(BB * TT / 64, 3 * HH);
  qkv_gemm_kernel<<<g1, 128, 0, stream>>>(Xbf, Wqkv, bq, bk, bv, Qbf, Kbf, Vcm);

  // Stage 2: attention.
  dim3 g2(TT / 16, HH, BB);
  const size_t smem_bytes = (size_t)(16 * TT + 8 * 16 * HDD + 256 + 16) * sizeof(float);
  attn_kernel<<<g2, 256, smem_bytes, stream>>>(Qbf, Kbf, Vcm, weights, Conc);

  // Stage 3: output projection.
  dim3 g3(BB * TT / 64, DD / 64);
  oproj_kernel<<<g3, 128, 0, stream>>>(Conc, WoCm, bo, out);
}
